// CustomTransformerLayer_48541720380029
// MI455X (gfx1250) — compile-verified, matrix operand order checked by reading, not run on hardware
//
#include <hip/hip_runtime.h>
#include <hip/hip_bf16.h>

typedef unsigned short u16;
typedef __bf16 v16bf __attribute__((ext_vector_type(16)));
typedef float v8f __attribute__((ext_vector_type(8)));

// ---------------------------------------------------------------------------
// helpers
// ---------------------------------------------------------------------------
__device__ __forceinline__ u16 f2bf(float f) {
    union { float f; unsigned u; } x;
    x.f = f;
    unsigned r = x.u + 0x7fffu + ((x.u >> 16) & 1u);   // round-to-nearest-even
    return (u16)(r >> 16);
}
__device__ __forceinline__ unsigned pack2(u16 lo, u16 hi) {
    return (unsigned)lo | ((unsigned)hi << 16);
}

// ---------------------------------------------------------------------------
// flat f32 -> bf16 convert (4 elems / thread)
// ---------------------------------------------------------------------------
__global__ __launch_bounds__(256) void cvt_f32_bf16(const float* __restrict__ in,
                                                    u16* __restrict__ out, size_t n4) {
    size_t i = (size_t)blockIdx.x * blockDim.x + threadIdx.x;
    if (i >= n4) return;
    float4 v = ((const float4*)in)[i];
    uint2 w;
    w.x = pack2(f2bf(v.x), f2bf(v.y));
    w.y = pack2(f2bf(v.z), f2bf(v.w));
    ((uint2*)out)[i] = w;
}

// ---------------------------------------------------------------------------
// transpose + convert: W[K][N] f32  ->  Wt[N][K] bf16   (K,N multiples of 32)
// block (32,8), grid (N/32, K/32)
// ---------------------------------------------------------------------------
__global__ __launch_bounds__(256) void tcvt_f32_bf16(const float* __restrict__ W,
                                                     u16* __restrict__ Wt, int K, int N) {
    __shared__ float tile[32][33];
    int n0 = blockIdx.x * 32, k0 = blockIdx.y * 32;
    int tx = threadIdx.x, ty = threadIdx.y;
#pragma unroll
    for (int i = 0; i < 32; i += 8)
        tile[ty + i][tx] = W[(size_t)(k0 + ty + i) * N + (n0 + tx)];
    __syncthreads();
#pragma unroll
    for (int i = 0; i < 32; i += 8)
        Wt[(size_t)(n0 + ty + i) * K + (k0 + tx)] = f2bf(tile[tx][ty + i]);
}

// ---------------------------------------------------------------------------
// bf16 batched transpose: V[b][S][D] -> Vt[b][D][S]
// block (32,8), grid (D/32, S/32, B)
// ---------------------------------------------------------------------------
__global__ __launch_bounds__(256) void transpose_bf16(const u16* __restrict__ V,
                                                      u16* __restrict__ Vt, int S, int D) {
    __shared__ u16 tile[32][34];
    size_t boff = (size_t)blockIdx.z * S * D;
    int d0 = blockIdx.x * 32, s0 = blockIdx.y * 32;
    int tx = threadIdx.x, ty = threadIdx.y;
#pragma unroll
    for (int i = 0; i < 32; i += 8)
        tile[ty + i][tx] = V[boff + (size_t)(s0 + ty + i) * D + (d0 + tx)];
    __syncthreads();
#pragma unroll
    for (int i = 0; i < 32; i += 8)
        Vt[boff + (size_t)(d0 + ty + i) * S + (s0 + tx)] = tile[tx][ty + i];
}

// ---------------------------------------------------------------------------
// WMMA bf16 GEMM (NT):  C[M][N] = A[M][K] * Bt[N][K]^T  (+bias, relu, resid)
// block = 256 threads (8 wave32), block tile 128x128, wave tile 32x64
// (2x4 = 8 WMMA per K32 step per wave), double-buffered LDS.
// Grid: (N/128, M/128, batches)
// ---------------------------------------------------------------------------
#define TM 128
#define TN 128
#define TK 32
#define LDST 40   // bf16 elems per LDS row (32 + 8 pad) -> 80B, 16B aligned

union Frag { uint4 u[2]; v16bf v; };

template<int BIAS, int RELU, int RESID, int OUTF32>
__global__ __launch_bounds__(256) void gemm_bf16_nt(
    const u16* __restrict__ A, const u16* __restrict__ Bt,
    const float* __restrict__ bias, const float* __restrict__ R,
    float* __restrict__ Cf, u16* __restrict__ Ch,
    int M, int N, int K, size_t strideA, size_t strideB, size_t strideC) {

    __shared__ u16 lds[2][(TM + TN) * LDST];

    const int tid  = threadIdx.x;
    const int lane = tid & 31;
    const int wave = tid >> 5;
    const int waveM = wave >> 1;        // 0..3 -> 32-row tile
    const int waveN = wave & 1;         // 0..1 -> 64-col tile
    const int half = lane >> 4;         // 0/1
    const int l16  = lane & 15;

    const size_t z = blockIdx.z;
    const u16* Ab = A + z * strideA + (size_t)(blockIdx.y * TM) * K;
    const u16* Bb = Bt + z * strideB + (size_t)(blockIdx.x * TN) * K;

    v8f acc[2][4];
#pragma unroll
    for (int mi = 0; mi < 2; mi++)
#pragma unroll
        for (int ni = 0; ni < 4; ni++)
#pragma unroll
            for (int r = 0; r < 8; r++) acc[mi][ni][r] = 0.0f;

    // ---- stage 0 direct load (A: 512 chunks, B: 512 chunks; 4/thread) ----
    {
#pragma unroll
        for (int i = 0; i < 2; i++) {
            int c = tid + i * 256;
            int row = c >> 2, seg = c & 3;
            *(uint4*)&lds[0][row * LDST + seg * 8] =
                *(const uint4*)(Ab + (size_t)row * K + seg * 8);
            *(uint4*)&lds[0][TM * LDST + row * LDST + seg * 8] =
                *(const uint4*)(Bb + (size_t)row * K + seg * 8);
        }
    }
    __syncthreads();

    const int nk = K / TK;
    for (int ks = 0; ks < nk; ks++) {
        const int buf = ks & 1;
        uint4 pa0, pa1, pb0, pb1;
        const bool more = (ks + 1 < nk);
        if (more) {
            size_t kb = (size_t)(ks + 1) * TK;
            {
                int c = tid, row = c >> 2, seg = c & 3;
                pa0 = *(const uint4*)(Ab + (size_t)row * K + kb + seg * 8);
                pb0 = *(const uint4*)(Bb + (size_t)row * K + kb + seg * 8);
            }
            {
                int c = tid + 256, row = c >> 2, seg = c & 3;
                pa1 = *(const uint4*)(Ab + (size_t)row * K + kb + seg * 8);
                pb1 = *(const uint4*)(Bb + (size_t)row * K + kb + seg * 8);
            }
        }

        // ---- fragments from LDS (ISA 7.12.2 bf16 A/B layout) ----
        const u16* lA = &lds[buf][0];
        const u16* lB = &lds[buf][TM * LDST];
        Frag a[2], b[4];
#pragma unroll
        for (int i = 0; i < 2; i++) {
            const u16* pA = lA + (waveM * 32 + i * 16 + l16) * LDST + half * 8;
            a[i].u[0] = *(const uint4*)pA;
            a[i].u[1] = *(const uint4*)(pA + 16);
        }
#pragma unroll
        for (int j = 0; j < 4; j++) {
            const u16* pB = lB + (waveN * 64 + j * 16 + l16) * LDST + half * 8;
            b[j].u[0] = *(const uint4*)pB;
            b[j].u[1] = *(const uint4*)(pB + 16);
        }

#pragma unroll
        for (int mi = 0; mi < 2; mi++)
#pragma unroll
            for (int ni = 0; ni < 4; ni++)
                acc[mi][ni] = __builtin_amdgcn_wmma_f32_16x16x32_bf16(
                    false, a[mi].v, false, b[ni].v, (short)0, acc[mi][ni],
                    false, false);

        if (more) {
            int c = tid, row = c >> 2, seg = c & 3;
            *(uint4*)&lds[buf ^ 1][row * LDST + seg * 8] = pa0;
            *(uint4*)&lds[buf ^ 1][TM * LDST + row * LDST + seg * 8] = pb0;
            c = tid + 256; row = c >> 2; seg = c & 3;
            *(uint4*)&lds[buf ^ 1][row * LDST + seg * 8] = pa1;
            *(uint4*)&lds[buf ^ 1][TM * LDST + row * LDST + seg * 8] = pb1;
        }
        __syncthreads();
    }

    // ---- epilogue ----
    const int rowBase = blockIdx.y * TM + waveM * 32;
    const int colBase = blockIdx.x * TN + waveN * 64;
    float* Cfb = OUTF32 ? (Cf + z * strideC) : nullptr;
    u16*   Chb = OUTF32 ? nullptr : (Ch + z * strideC);
    const float* Rb = RESID ? (R + z * strideC) : nullptr;

#pragma unroll
    for (int mi = 0; mi < 2; mi++) {
#pragma unroll
        for (int ni = 0; ni < 4; ni++) {
            int col = colBase + ni * 16 + l16;
            float bi = BIAS ? bias[col] : 0.0f;
#pragma unroll
            for (int r = 0; r < 8; r++) {
                int row = rowBase + mi * 16 + half * 8 + r;
                float v = acc[mi][ni][r] + bi;
                if (RELU) v = fmaxf(v, 0.0f);
                size_t idx = (size_t)row * N + col;
                if (RESID) v += Rb[idx];
                if (OUTF32) Cfb[idx] = v;
                else        Chb[idx] = f2bf(v);
            }
        }
    }
}

// ---------------------------------------------------------------------------
// row softmax: in f32 [rows][2048] -> out bf16.  One 256-thread block per row.
// ---------------------------------------------------------------------------
__global__ __launch_bounds__(256) void softmax_rows(const float* __restrict__ in,
                                                    u16* __restrict__ out, int len) {
    __shared__ float red[8];
    const size_t row = blockIdx.x;
    const int t = threadIdx.x;
    const float* p = in + row * (size_t)len + t * 8;

    float4 v0 = *(const float4*)(p);
    float4 v1 = *(const float4*)(p + 4);
    float m = fmaxf(fmaxf(fmaxf(v0.x, v0.y), fmaxf(v0.z, v0.w)),
                    fmaxf(fmaxf(v1.x, v1.y), fmaxf(v1.z, v1.w)));
#pragma unroll
    for (int o = 16; o; o >>= 1) m = fmaxf(m, __shfl_xor(m, o, 32));
    if ((t & 31) == 0) red[t >> 5] = m;
    __syncthreads();
    m = red[0];
#pragma unroll
    for (int i = 1; i < 8; i++) m = fmaxf(m, red[i]);
    __syncthreads();

    float e[8];
    e[0] = __expf(v0.x - m); e[1] = __expf(v0.y - m);
    e[2] = __expf(v0.z - m); e[3] = __expf(v0.w - m);
    e[4] = __expf(v1.x - m); e[5] = __expf(v1.y - m);
    e[6] = __expf(v1.z - m); e[7] = __expf(v1.w - m);
    float s = 0.0f;
#pragma unroll
    for (int i = 0; i < 8; i++) s += e[i];
#pragma unroll
    for (int o = 16; o; o >>= 1) s += __shfl_xor(s, o, 32);
    if ((t & 31) == 0) red[t >> 5] = s;
    __syncthreads();
    s = 0.0f;
#pragma unroll
    for (int i = 0; i < 8; i++) s += red[i];
    float inv = 1.0f / s;

    uint4 w;
    w.x = pack2(f2bf(e[0] * inv), f2bf(e[1] * inv));
    w.y = pack2(f2bf(e[2] * inv), f2bf(e[3] * inv));
    w.z = pack2(f2bf(e[4] * inv), f2bf(e[5] * inv));
    w.w = pack2(f2bf(e[6] * inv), f2bf(e[7] * inv));
    *(uint4*)(out + row * (size_t)len + t * 8) = w;
}

// ---------------------------------------------------------------------------
// LayerNorm: in f32 [rows][1024]; writes f32 (and optional bf16) normalized.
// One 256-thread block per row (4 elems / thread).
// ---------------------------------------------------------------------------
__global__ __launch_bounds__(256) void layernorm_rows(const float* __restrict__ in,
                                                      const float* __restrict__ g,
                                                      const float* __restrict__ bta,
                                                      float* __restrict__ fout,
                                                      u16* __restrict__ hout) {
    __shared__ float red[16];
    const size_t row = blockIdx.x;
    const int t = threadIdx.x;
    const float* p = in + row * 1024 + t * 4;
    float4 v = *(const float4*)p;
    float s = v.x + v.y + v.z + v.w;
    float q = v.x * v.x + v.y * v.y + v.z * v.z + v.w * v.w;
#pragma unroll
    for (int o = 16; o; o >>= 1) { s += __shfl_xor(s, o, 32); q += __shfl_xor(q, o, 32); }
    if ((t & 31) == 0) { red[t >> 5] = s; red[8 + (t >> 5)] = q; }
    __syncthreads();
    s = 0.0f; q = 0.0f;
#pragma unroll
    for (int i = 0; i < 8; i++) { s += red[i]; q += red[8 + i]; }
    float mean = s * (1.0f / 1024.0f);
    float var = q * (1.0f / 1024.0f) - mean * mean;
    float rs = rsqrtf(var + 1e-5f);

    int c = t * 4;
    float4 o;
    o.x = (v.x - mean) * rs * g[c + 0] + bta[c + 0];
    o.y = (v.y - mean) * rs * g[c + 1] + bta[c + 1];
    o.z = (v.z - mean) * rs * g[c + 2] + bta[c + 2];
    o.w = (v.w - mean) * rs * g[c + 3] + bta[c + 3];
    *(float4*)(fout + row * 1024 + c) = o;
    if (hout) {
        uint2 w;
        w.x = pack2(f2bf(o.x), f2bf(o.y));
        w.y = pack2(f2bf(o.z), f2bf(o.w));
        *(uint2*)(hout + row * 1024 + c) = w;
    }
}

// ---------------------------------------------------------------------------
// host: orchestrate the whole layer
// ---------------------------------------------------------------------------
extern "C" void kernel_launch(void* const* d_in, const int* in_sizes, int n_in,
                              void* d_out, int out_size, void* d_ws, size_t ws_size,
                              hipStream_t stream) {
    constexpr int B = 4, S = 2048, D = 1024, F = 4096;
    constexpr int BS = B * S;

    const float* src = (const float*)d_in[0];
    const float* Wq = (const float*)d_in[1];
    const float* bq = (const float*)d_in[2];
    const float* Wk = (const float*)d_in[3];
    const float* bk = (const float*)d_in[4];
    const float* Wv = (const float*)d_in[5];
    const float* bv = (const float*)d_in[6];
    const float* Wo = (const float*)d_in[7];
    const float* bo = (const float*)d_in[8];
    const float* W1 = (const float*)d_in[9];
    const float* b1 = (const float*)d_in[10];
    const float* W2 = (const float*)d_in[11];
    const float* b2 = (const float*)d_in[12];
    const float* g1 = (const float*)d_in[13];
    const float* be1 = (const float*)d_in[14];
    const float* g2 = (const float*)d_in[15];
    const float* be2 = (const float*)d_in[16];

    char* ws = (char*)d_ws;
    const size_t MB = 1ull << 20;
    u16* X0h   = (u16*)(ws + 0);          // 16MB
    u16* WqT   = (u16*)(ws + 16 * MB);    //  2MB
    u16* WkT   = (u16*)(ws + 18 * MB);    //  2MB
    u16* WvT   = (u16*)(ws + 20 * MB);    //  2MB
    u16* WoT   = (u16*)(ws + 22 * MB);    //  2MB
    u16* W1T   = (u16*)(ws + 24 * MB);    //  8MB
    u16* W2T   = (u16*)(ws + 32 * MB);    //  8MB
    u16* Qh    = (u16*)(ws + 40 * MB);    // 16MB
    u16* Kh    = (u16*)(ws + 56 * MB);    // 16MB
    u16* Vh    = (u16*)(ws + 72 * MB);    // 16MB
    u16* Vt    = (u16*)(ws + 88 * MB);    // 16MB
    u16* attnh = (u16*)(ws + 104 * MB);   // 16MB
    float* scores = (float*)(ws + 120 * MB);  // 64MB
    u16* Ph    = (u16*)(ws + 184 * MB);   // 32MB
    // aliases (non-overlapping lifetimes):
    float* y1  = (float*)(ws + 184 * MB); // over Ph   (32MB)
    float* x1  = (float*)(ws + 40 * MB);  // over Qh+Kh (32MB)
    u16*  x1h  = (u16*)(ws + 72 * MB);    // over Vh   (16MB)
    u16*  ffh  = (u16*)(ws + 120 * MB);   // over scores (64MB)
    float* y2  = (float*)(ws + 88 * MB);  // over Vt+attnh (32MB)

    dim3 tblk(32, 8);

    // 1. conversions
    cvt_f32_bf16<<<(BS * D / 4) / 256, 256, 0, stream>>>(src, X0h, (size_t)BS * D / 4);
    tcvt_f32_bf16<<<dim3(D / 32, D / 32), tblk, 0, stream>>>(Wq, WqT, D, D);
    tcvt_f32_bf16<<<dim3(D / 32, D / 32), tblk, 0, stream>>>(Wk, WkT, D, D);
    tcvt_f32_bf16<<<dim3(D / 32, D / 32), tblk, 0, stream>>>(Wv, WvT, D, D);
    tcvt_f32_bf16<<<dim3(D / 32, D / 32), tblk, 0, stream>>>(Wo, WoT, D, D);
    tcvt_f32_bf16<<<dim3(F / 32, D / 32), tblk, 0, stream>>>(W1, W1T, D, F);
    tcvt_f32_bf16<<<dim3(D / 32, F / 32), tblk, 0, stream>>>(W2, W2T, F, D);

    // 2. Q/K/V projections: [BS x D] = X0h * W^T + b    (bf16 out)
    gemm_bf16_nt<1, 0, 0, 0><<<dim3(D / TN, BS / TM, 1), 256, 0, stream>>>(
        X0h, WqT, bq, nullptr, nullptr, Qh, BS, D, D, 0, 0, 0);
    gemm_bf16_nt<1, 0, 0, 0><<<dim3(D / TN, BS / TM, 1), 256, 0, stream>>>(
        X0h, WkT, bk, nullptr, nullptr, Kh, BS, D, D, 0, 0, 0);
    gemm_bf16_nt<1, 0, 0, 0><<<dim3(D / TN, BS / TM, 1), 256, 0, stream>>>(
        X0h, WvT, bv, nullptr, nullptr, Vh, BS, D, D, 0, 0, 0);

    // 3. V -> Vt [B][D][S]
    transpose_bf16<<<dim3(D / 32, S / 32, B), tblk, 0, stream>>>(Vh, Vt, S, D);

    // 4. scores[b] = Q_b * K_b^T  (f32 out), batched
    gemm_bf16_nt<0, 0, 0, 1><<<dim3(S / TN, S / TM, B), 256, 0, stream>>>(
        Qh, Kh, nullptr, nullptr, scores, nullptr, S, S, D,
        (size_t)S * D, (size_t)S * D, (size_t)S * S);

    // 5. softmax rows -> bf16 probabilities
    softmax_rows<<<B * S, 256, 0, stream>>>(scores, Ph, S);

    // 6. attn[b] = P_b * V_b  == P_b * Vt_b^T  (bf16 out), batched
    gemm_bf16_nt<0, 0, 0, 0><<<dim3(D / TN, S / TM, B), 256, 0, stream>>>(
        Ph, Vt, nullptr, nullptr, nullptr, attnh, S, D, S,
        (size_t)S * S, (size_t)D * S, (size_t)S * D);

    // 7. y1 = attn * Wo^T + bo + src   (f32 out)
    gemm_bf16_nt<1, 0, 1, 1><<<dim3(D / TN, BS / TM, 1), 256, 0, stream>>>(
        attnh, WoT, bo, src, y1, nullptr, BS, D, D, 0, 0, 0);

    // 8. x1 = LN(y1)  (f32 + bf16)
    layernorm_rows<<<BS, 256, 0, stream>>>(y1, g1, be1, x1, x1h);

    // 9. ffh = relu(x1 * W1^T + b1)  (bf16 out)
    gemm_bf16_nt<1, 1, 0, 0><<<dim3(F / TN, BS / TM, 1), 256, 0, stream>>>(
        x1h, W1T, b1, nullptr, nullptr, ffh, BS, F, D, 0, 0, 0);

    // 10. y2 = ffh * W2^T + b2 + x1  (f32 out)
    gemm_bf16_nt<1, 0, 1, 1><<<dim3(D / TN, BS / TM, 1), 256, 0, stream>>>(
        ffh, W2T, b2, x1, y2, nullptr, BS, D, F, 0, 0, 0);

    // 11. out = LN(y2)
    layernorm_rows<<<BS, 256, 0, stream>>>(y2, g2, be2, (float*)d_out, nullptr);

    (void)in_sizes; (void)n_in; (void)out_size; (void)ws_size;
}
